// RTM_23811298689122
// MI455X (gfx1250) — matrix operand order repaired and checked by reading, not compiled, not run on hardware
//
#include <hip/hip_runtime.h>
#include <stdint.h>

#define EPS 1e-5f
#define LDS_SROW 776   // 768 + 8 f32 pad (row stride 3104 B, 16B aligned)
#define LDS_BROW 784   // 768 + 16 bf16 pad (row stride 1568 B, 16B aligned)

typedef __attribute__((ext_vector_type(16))) __bf16 v16bf;
typedef __attribute__((ext_vector_type(8)))  float  v8f;

union ABfrag { uint4 q[2]; uint32_t u[8]; v16bf v; };

__device__ __forceinline__ uint16_t f2bf_sw(float f) {
  union { float f; uint32_t u; } c; c.f = f;
  uint32_t u = c.u;
  return (uint16_t)((u + 0x7FFFu + ((u >> 16) & 1u)) >> 16);   // RNE
}
__device__ __forceinline__ uint32_t pk_bf16(float a, float b) {
#if __has_builtin(__builtin_amdgcn_cvt_pk_bf16_f32)
  typedef __attribute__((ext_vector_type(2))) __bf16 v2bf;
  union { v2bf v; uint32_t u; } c;
  c.v = __builtin_amdgcn_cvt_pk_bf16_f32(a, b);
  return c.u;
#else
  return (uint32_t)f2bf_sw(a) | ((uint32_t)f2bf_sw(b) << 16);
#endif
}
__device__ __forceinline__ uint16_t f2bf(float a) { return (uint16_t)(pk_bf16(a, a) & 0xFFFFu); }

__device__ __forceinline__ float wave_sum(float v) {
  #pragma unroll
  for (int o = 16; o > 0; o >>= 1) v += __shfl_xor(v, o, 32);
  return v;
}
__device__ __forceinline__ v8f wmma_bf16(const ABfrag& A, const ABfrag& B, v8f C) {
  return __builtin_amdgcn_wmma_f32_16x16x32_bf16(false, A.v, false, B.v, (short)0, C, false, false);
}

// ---------------- K1: f32 -> bf16 conversion (pairwise, packed cvt) ----------------
__global__ void k_cvt_bf16(const float* __restrict__ src, uint32_t* __restrict__ dst, int npair) {
  int i = blockIdx.x * blockDim.x + threadIdx.x;
  int st = gridDim.x * blockDim.x;
  const float2* s2 = (const float2*)src;
  for (; i < npair; i += st) { float2 v = s2[i]; dst[i] = pk_bf16(v.x, v.y); }
}

// ---------------- K1b: positional-bias table PE[256][768] ----------------
__global__ void k_pe(float* __restrict__ pe) {
  int idx = blockIdx.x * blockDim.x + threadIdx.x;
  if (idx >= 256 * 768) return;
  int j = idx / 768, m = idx % 768;
  int mz = m & ~1;
  float inv = expf((float)mz * (-9.210340371976184f / 384.0f));  // 10000^(-mz/384)
  float ang = (float)j * inv;
  pe[idx] = (m & 1) ? cosf(ang) : sinf(ang);
}

// ---------------- K2: s0 = x @ weight^T + PE  (4096x768, K=768, all-bf16 operands) ----------------
__global__ __launch_bounds__(256) void k_s0_gemm(const uint16_t* __restrict__ xb,
                                                 const uint16_t* __restrict__ Wbf,
                                                 const float* __restrict__ pe,
                                                 float* __restrict__ s0) {
  int gw   = (blockIdx.x * blockDim.x + threadIdx.x) >> 5;
  int lane = threadIdx.x & 31;
  int rt = gw / 48, ct = gw % 48;
  if (rt >= 256) return;
  int l16 = lane & 15, h = lane >> 4;
  const uint16_t* ap = xb  + (size_t)(rt * 16 + l16) * 768;
  const uint16_t* bp = Wbf + (size_t)(ct * 16 + l16) * 768 + 16 * h;
  v8f acc = {};
  for (int kt = 0; kt < 24; ++kt) {
    int k0 = kt * 32;
    ABfrag A, B;
    A.q[0] = *(const uint4*)(ap + k0 + 16 * h);
    A.q[1] = *(const uint4*)(ap + k0 + 16 * h + 8);
    const uint4* bq = (const uint4*)(bp + k0);
    B.q[0] = bq[0]; B.q[1] = bq[1];
    acc = wmma_bf16(A, B, acc);
  }
  #pragma unroll
  for (int i = 0; i < 8; ++i) {
    int row = rt * 16 + i + 8 * h;
    int m   = ct * 16 + l16;
    s0[(size_t)row * 768 + m] = acc[i] + pe[(row & 255) * 768 + m]; // row = batch*256 + token
  }
}

// ---------------- flag init ----------------
__global__ void k_init_flags(int* flags) {
  if (threadIdx.x < 16) flags[threadIdx.x] = 0;
}

// ---------------- K3: 10-stage pipelined causal scan ----------------
__global__ __launch_bounds__(512) void k_scan(
    const float* __restrict__ s0,
    const uint16_t* __restrict__ Wv_bf, const uint16_t* __restrict__ Wo_bf,
    const uint16_t* __restrict__ F1bf,  const uint16_t* __restrict__ F2bf,
    const float* __restrict__ g1, const float* __restrict__ b1,
    const float* __restrict__ g2, const float* __restrict__ b2,
    const float* __restrict__ fb1, const float* __restrict__ fb2,
    float* __restrict__ cumV, float* __restrict__ pipes, int* __restrict__ flags,
    float* __restrict__ out) {

  __shared__ __align__(16) float    s_lds[16 * LDS_SROW];
  __shared__ __align__(16) uint16_t snb  [16 * LDS_BROW];
  __shared__ __align__(16) uint16_t bufB [16 * LDS_BROW];

  const int a    = blockIdx.x;          // stage this workgroup owns
  const int tid  = threadIdx.x;
  const int lane = tid & 31;
  const int w    = tid >> 5;            // wave 0..15
  const int l16  = lane & 15, h = lane >> 4;

  const uint16_t* Wv = Wv_bf + (size_t)a * 768 * 768;
  const uint16_t* Wo = Wo_bf + (size_t)a * 768 * 768;
  float* cv = cumV + (size_t)a * 16 * 768;

  for (int idx = tid; idx < 16 * 768; idx += 512) cv[idx] = 0.0f;  // cumV0 = 0
  __syncthreads();

  for (int j = 0; j < 256; ++j) {
    // producer wait + slot back-pressure
    if (tid == 0) {
      if (a > 0)
        while (__hip_atomic_load(&flags[a - 1], __ATOMIC_ACQUIRE, __HIP_MEMORY_SCOPE_AGENT) < j + 1)
          __builtin_amdgcn_s_sleep(2);
      if (a < 9 && j >= 2)
        while (__hip_atomic_load(&flags[a + 1], __ATOMIC_ACQUIRE, __HIP_MEMORY_SCOPE_AGENT) < j - 1)
          __builtin_amdgcn_s_sleep(2);
    }
    __syncthreads();
    __threadfence();   // agent acquire: make producer's data visible to whole WG

    // load token state
    if (a == 0) {
      for (int idx = tid; idx < 16 * 768; idx += 512) {
        int i = idx / 768, m = idx % 768;
        s_lds[i * LDS_SROW + m] = s0[(size_t)(i * 256 + j) * 768 + m];
      }
    } else {
      const float* pin = pipes + (size_t)((a - 1) * 2 + ((j + a) & 1)) * 12288;
      for (int idx = tid; idx < 16 * 768; idx += 512)
        s_lds[(idx / 768) * LDS_SROW + (idx % 768)] = pin[idx];
    }
    __syncthreads();

    // ---- LN1 -> snb (bf16); one wave per row, adjacent pairs per lane ----
    {
      float p = 0.f;
      for (int k = 2 * lane; k < 768; k += 64) p += s_lds[w * LDS_SROW + k] + s_lds[w * LDS_SROW + k + 1];
      float mu = wave_sum(p) * (1.0f / 768.0f);
      float q = 0.f;
      for (int k = 2 * lane; k < 768; k += 64) {
        float d0 = s_lds[w * LDS_SROW + k] - mu, d1 = s_lds[w * LDS_SROW + k + 1] - mu;
        q += d0 * d0 + d1 * d1;
      }
      float rstd = rsqrtf(wave_sum(q) * (1.0f / 768.0f) + EPS);
      for (int k = 2 * lane; k < 768; k += 64) {
        float v0 = (s_lds[w * LDS_SROW + k]     - mu) * rstd * g1[k]     + b1[k];
        float v1 = (s_lds[w * LDS_SROW + k + 1] - mu) * rstd * g1[k + 1] + b1[k + 1];
        *(uint32_t*)&snb[w * LDS_BROW + k] = pk_bf16(v0, v1);
      }
    }
    // imv (old cumV) -> bufB bf16 (pairwise)
    for (int idx = tid; idx < 16 * 384; idx += 512) {
      int i = idx / 384, m2 = (idx % 384) * 2;
      float2 v = *(const float2*)(cv + i * 768 + m2);
      *(uint32_t*)&bufB[i * LDS_BROW + m2] = pk_bf16(v.x, v.y);
    }
    __syncthreads();

    // ---- v = LN1(s) @ Wv^T  (softmax of scalar == 1 -> cumV += v) ; s += Wo @ imv ----
    for (int t = 0; t < 3; ++t) {
      int ct = w + 16 * t;
      v8f acc = {};
      const uint16_t* bp = Wv + (size_t)(ct * 16 + l16) * 768 + 16 * h;
      for (int kt = 0; kt < 24; ++kt) {
        ABfrag A, B;
        A.q[0] = *(const uint4*)&snb[l16 * LDS_BROW + kt * 32 + 16 * h];
        A.q[1] = *(const uint4*)&snb[l16 * LDS_BROW + kt * 32 + 16 * h + 8];
        const uint4* bq = (const uint4*)(bp + kt * 32);
        B.q[0] = bq[0]; B.q[1] = bq[1];
        acc = wmma_bf16(A, B, acc);
      }
      #pragma unroll
      for (int i = 0; i < 8; ++i) {
        float* p = cv + (i + 8 * h) * 768 + ct * 16 + l16;
        *p = *p + acc[i];
      }
      v8f acc2 = {};
      const uint16_t* bp2 = Wo + (size_t)(ct * 16 + l16) * 768 + 16 * h;
      for (int kt = 0; kt < 24; ++kt) {
        ABfrag A, B;
        A.q[0] = *(const uint4*)&bufB[l16 * LDS_BROW + kt * 32 + 16 * h];
        A.q[1] = *(const uint4*)&bufB[l16 * LDS_BROW + kt * 32 + 16 * h + 8];
        const uint4* bq = (const uint4*)(bp2 + kt * 32);
        B.q[0] = bq[0]; B.q[1] = bq[1];
        acc2 = wmma_bf16(A, B, acc2);
      }
      #pragma unroll
      for (int i = 0; i < 8; ++i)
        s_lds[(i + 8 * h) * LDS_SROW + ct * 16 + l16] += acc2[i];
    }
    __syncthreads();

    // ---- s = LN2(s) + s -> snb (mlp input) ----
    {
      float p = 0.f;
      for (int k = 2 * lane; k < 768; k += 64) p += s_lds[w * LDS_SROW + k] + s_lds[w * LDS_SROW + k + 1];
      float mu = wave_sum(p) * (1.0f / 768.0f);
      float q = 0.f;
      for (int k = 2 * lane; k < 768; k += 64) {
        float d0 = s_lds[w * LDS_SROW + k] - mu, d1 = s_lds[w * LDS_SROW + k + 1] - mu;
        q += d0 * d0 + d1 * d1;
      }
      float rstd = rsqrtf(wave_sum(q) * (1.0f / 768.0f) + EPS);
      for (int k = 2 * lane; k < 768; k += 64) {
        float x0 = s_lds[w * LDS_SROW + k], x1 = s_lds[w * LDS_SROW + k + 1];
        float t0 = (x0 - mu) * rstd * g2[k]     + b2[k]     + x0;
        float t1 = (x1 - mu) * rstd * g2[k + 1] + b2[k + 1] + x1;
        *(uint32_t*)&snb[w * LDS_BROW + k] = pk_bf16(t0, t1);
      }
    }
    __syncthreads();

    // ---- MLP: fc1(gelu) / fc2, chunked over K=3072 in 4 chunks of 768 ----
    v8f oacc[3] = {{}, {}, {}};
    for (int c = 0; c < 4; ++c) {
      for (int t = 0; t < 3; ++t) {
        int ctl = w + 16 * t;
        int feat = c * 768 + ctl * 16 + l16;              // fc1 output feature (B column)
        v8f acc = {};
        const uint16_t* bp = F1bf + (size_t)feat * 768 + 16 * h;
        for (int kt = 0; kt < 24; ++kt) {
          ABfrag A, B;
          A.q[0] = *(const uint4*)&snb[l16 * LDS_BROW + kt * 32 + 16 * h];
          A.q[1] = *(const uint4*)&snb[l16 * LDS_BROW + kt * 32 + 16 * h + 8];
          const uint4* bq = (const uint4*)(bp + kt * 32);
          B.q[0] = bq[0]; B.q[1] = bq[1];
          acc = wmma_bf16(A, B, acc);
        }
        #pragma unroll
        for (int i = 0; i < 8; ++i) {
          float xv = acc[i] + fb1[feat];
          float ge = 0.5f * xv * (1.0f + erff(xv * 0.70710678118654752f)); // exact gelu
          bufB[(i + 8 * h) * LDS_BROW + ctl * 16 + l16] = f2bf(ge);
        }
      }
      __syncthreads();
      #pragma unroll
      for (int t = 0; t < 3; ++t) {
        int ot = w + 16 * t;
        const uint16_t* bp = F2bf + (size_t)(ot * 16 + l16) * 3072 + c * 768 + 16 * h;
        for (int kt = 0; kt < 24; ++kt) {
          ABfrag A, B;
          A.q[0] = *(const uint4*)&bufB[l16 * LDS_BROW + kt * 32 + 16 * h];
          A.q[1] = *(const uint4*)&bufB[l16 * LDS_BROW + kt * 32 + 16 * h + 8];
          const uint4* bq = (const uint4*)(bp + kt * 32);
          B.q[0] = bq[0]; B.q[1] = bq[1];
          oacc[t] = wmma_bf16(A, B, oacc[t]);
        }
      }
      __syncthreads();
    }
    #pragma unroll
    for (int t = 0; t < 3; ++t) {
      int col = (w + 16 * t) * 16 + l16;
      #pragma unroll
      for (int i = 0; i < 8; ++i)
        s_lds[(i + 8 * h) * LDS_SROW + col] = oacc[t][i] + fb2[col];
    }
    __syncthreads();

    // emit
    if (a == 9) {
      for (int idx = tid; idx < 16 * 768; idx += 512) {
        int i = idx / 768, m = idx % 768;
        out[(size_t)(i * 256 + j) * 768 + m] = s_lds[i * LDS_SROW + m];
      }
    } else {
      float* pout = pipes + (size_t)(a * 2 + ((j + a + 1) & 1)) * 12288;
      for (int idx = tid; idx < 16 * 768; idx += 512)
        pout[idx] = s_lds[(idx / 768) * LDS_SROW + (idx % 768)];
    }
    __threadfence();   // agent release of data
    __syncthreads();
    if (tid == 0)
      __hip_atomic_store(&flags[a], j + 1, __ATOMIC_RELEASE, __HIP_MEMORY_SCOPE_AGENT);
  }
}

extern "C" void kernel_launch(void* const* d_in, const int* in_sizes, int n_in,
                              void* d_out, int out_size, void* d_ws, size_t ws_size,
                              hipStream_t stream) {
  (void)in_sizes; (void)n_in; (void)out_size; (void)ws_size;
  const float* x      = (const float*)d_in[0];
  const float* weight = (const float*)d_in[1];
  const float* Wv     = (const float*)d_in[4];   // Wq (2) / Wk (3) are dead: softmax of a scalar == 1
  const float* Wo     = (const float*)d_in[5];
  const float* ln1_g  = (const float*)d_in[6];
  const float* ln1_b  = (const float*)d_in[7];
  const float* ln2_g  = (const float*)d_in[8];
  const float* ln2_b  = (const float*)d_in[9];
  const float* fc1_w  = (const float*)d_in[10];
  const float* fc1_b  = (const float*)d_in[11];
  const float* fc2_w  = (const float*)d_in[12];
  const float* fc2_b  = (const float*)d_in[13];
  float* out = (float*)d_out;

  uint8_t* ws = (uint8_t*)d_ws;
  const size_t o_s0 = 0;                       // 4096*768 f32
  const size_t o_w  = o_s0 + 12582912;         // weight bf16
  const size_t o_wv = o_w  + 1179648;          // 10x768x768 bf16
  const size_t o_wo = o_wv + 11796480;
  const size_t o_f1 = o_wo + 11796480;         // 3072x768 bf16
  const size_t o_f2 = o_f1 + 4718592;          // 768x3072 bf16
  const size_t o_xb = o_f2 + 4718592;          // 4096x768 bf16 (x)
  const size_t o_pe = o_xb + 6291456;          // 256x768 f32 PE table
  const size_t o_cv = o_pe + 786432;           // 10x16x768 f32
  const size_t o_pp = o_cv + 491520;           // 10x2x16x768 f32 pipes
  const size_t o_fl = o_pp + 983040;           // flags

  uint16_t* w_bf  = (uint16_t*)(ws + o_w);
  uint16_t* wv_bf = (uint16_t*)(ws + o_wv);
  uint16_t* wo_bf = (uint16_t*)(ws + o_wo);
  uint16_t* f1_bf = (uint16_t*)(ws + o_f1);
  uint16_t* f2_bf = (uint16_t*)(ws + o_f2);
  uint16_t* x_bf  = (uint16_t*)(ws + o_xb);
  float* s0    = (float*)(ws + o_s0);
  float* pe    = (float*)(ws + o_pe);
  float* cumV  = (float*)(ws + o_cv);
  float* pipes = (float*)(ws + o_pp);
  int*   flags = (int*)(ws + o_fl);

  auto cvt = [&](const float* s, uint16_t* d, int n) {
    int npair = n / 2;
    int blk = (npair + 255) / 256; if (blk > 4096) blk = 4096;
    k_cvt_bf16<<<blk, 256, 0, stream>>>(s, (uint32_t*)d, npair);
  };
  cvt(weight, w_bf, 768 * 768);
  cvt(Wv,     wv_bf, 10 * 768 * 768);
  cvt(Wo,     wo_bf, 10 * 768 * 768);
  cvt(fc1_w,  f1_bf, 3072 * 768);
  cvt(fc2_w,  f2_bf, 768 * 3072);
  cvt(x,      x_bf,  4096 * 768);
  k_pe<<<(256 * 768 + 255) / 256, 256, 0, stream>>>(pe);

  k_s0_gemm<<<1536, 256, 0, stream>>>(x_bf, w_bf, pe, s0);   // 256*48 wave tiles
  k_init_flags<<<1, 32, 0, stream>>>(flags);
  k_scan<<<10, 512, 0, stream>>>(s0, wv_bf, wo_bf, f1_bf, f2_bf,
                                 ln1_g, ln1_b, ln2_g, ln2_b, fc1_b, fc2_b,
                                 cumV, pipes, flags, out);
}
